// CCInitPi_86277303042196
// MI455X (gfx1250) — compile-verified, matrix operand order checked by reading, not run on hardware
//
#include <hip/hip_runtime.h>

// D=128, N=8192, K=256
#define DD   128
#define NN   8192
#define KK   256
#define PITCH 136   // 128 bf16 + 8 pad -> 272B row, 16B-aligned, bank-spread

typedef __bf16 v16bf __attribute__((ext_vector_type(16)));
typedef float  v8f   __attribute__((ext_vector_type(8)));
typedef unsigned int v4u __attribute__((ext_vector_type(4)));
typedef int v4i __attribute__((ext_vector_type(4)));
typedef int v8i __attribute__((ext_vector_type(8)));

union Frag { uint4 u[2]; v16bf v; };

// ---------------------------------------------------------------------------
// Tensor Data Mover: async 2D tile load (rows x 128 bf16) global -> LDS with
// hardware row padding: 256B row + 16B pad  (pad_interval=5 -> 64 DWORDs,
// pad_amount=3 -> 4 DWORDs) == PITCH=136 elements. D# per ISA 08 §8.
// ---------------------------------------------------------------------------
__device__ inline void tdm_load_tile(const __bf16* gsrc, unsigned ldsOff, int rows) {
    const unsigned long long ga = (unsigned long long)(uintptr_t)gsrc;
    v4u g0;
    g0.x = 1u;                                   // count=1, user descriptor
    g0.y = ldsOff;                               // lds_addr
    g0.z = (unsigned)ga;                         // global_addr[31:0]
    g0.w = (unsigned)((ga >> 32) & 0x01FFFFFFu)  // global_addr[56:32]
         | (2u << 30);                           // type=2 ("image")
    v8i g1;
    g1[0] = (int)((1u << 16)      // data_size = 1 -> 2 bytes (bf16)
                | (1u << 20)      // pad_enable
                | (5u << 22)      // pad_interval: 64 DWORDs (256B) per row
                | (3u << 25));    // pad_amount: 4 DWORDs (16B)
    g1[1] = (int)(128u << 16);    // tensor_dim0 = 128 (elements per row)
    g1[2] = (int)((unsigned)rows << 16);   // tensor_dim1 = rows
    g1[3] = (int)(128u << 16);    // tile_dim0 = 128
    g1[4] = rows;                 // tile_dim1 = rows, tile_dim2 = 0
    g1[5] = 128;                  // tensor_dim0_stride = 128 elements
    g1[6] = 0;
    g1[7] = 0;
    v4i z4 = {0, 0, 0, 0};
#if __clang_major__ >= 23
    v8i z8 = {0, 0, 0, 0, 0, 0, 0, 0};
    __builtin_amdgcn_tensor_load_to_lds(g0, g1, z4, z4, z8, 0);
#else
    __builtin_amdgcn_tensor_load_to_lds(g0, g1, z4, z4, 0);
#endif
}

// ---------------------------------------------------------------------------
// Tiled transpose + fp32->bf16 convert: out[c*R + r] = (bf16) in[r*C + c]
// grid (C/32, R/32), block (32,8)
// ---------------------------------------------------------------------------
__global__ __launch_bounds__(256)
void transpose_cvt_kernel(const float* __restrict__ in, __bf16* __restrict__ out,
                          int R, int C) {
    __shared__ float tile[32][33];
    const int rBase = blockIdx.y * 32, cBase = blockIdx.x * 32;
    const int tx = threadIdx.x, ty = threadIdx.y;
#pragma unroll
    for (int i = 0; i < 32; i += 8)
        tile[ty + i][tx] = in[(size_t)(rBase + ty + i) * C + cBase + tx];
    __syncthreads();
#pragma unroll
    for (int i = 0; i < 32; i += 8)
        out[(size_t)(cBase + ty + i) * R + rBase + tx] = (__bf16)tile[tx][ty + i];
}

// ---------------------------------------------------------------------------
// Amu[c*128 + d] = sum_e N_A[d,e,c] * N_mu[e,c]   (fp32, tiny)
// ---------------------------------------------------------------------------
__global__ __launch_bounds__(256)
void amu_kernel(const float* __restrict__ NA, const float* __restrict__ Nmu,
                float* __restrict__ Amu) {
    const int c = threadIdx.x;
    const int d = blockIdx.x;
    float acc = 0.f;
#pragma unroll 4
    for (int e = 0; e < DD; ++e)
        acc += NA[((size_t)d * DD + e) * KK + c] * Nmu[(size_t)e * KK + c];
    Amu[(size_t)c * DD + d] = acc;
}

// ---------------------------------------------------------------------------
// Per-c compute: full d-reduction of (A_c X - Amu_c)^2 for this wave's 32 n.
// Software-pipelined A-fragment loads (register double buffer).
// ---------------------------------------------------------------------------
__device__ __forceinline__ void load_frag(Frag& f, const __bf16* p, int hi) {
    f.u[0] = *(const uint4*)(p + hi * 8);
    f.u[1] = *(const uint4*)(p + 16 + hi * 8);
}

__device__ __forceinline__ void compute_c(const __bf16* __restrict__ lAb,
                                          const float* __restrict__ amub,
                                          const Frag (&bfrag)[2][4],
                                          int lo, int hi,
                                          float& sq0o, float& sq1o) {
    float sq0 = 0.f, sq1 = 0.f;
    Frag af[2];
    load_frag(af[0], &lAb[lo * PITCH], hi);
    v8f acc0, acc1;
#pragma unroll
    for (int step = 0; step < 32; ++step) {
        const int dc = step >> 2, ec = step & 3;
        if (step + 1 < 32) {   // prefetch next A fragment into the other buffer
            const int dn = (step + 1) >> 2, en = (step + 1) & 3;
            load_frag(af[(step + 1) & 1], &lAb[(dn * 16 + lo) * PITCH + en * 32], hi);
        }
        if (ec == 0) {
            acc0 = (v8f){0.f, 0.f, 0.f, 0.f, 0.f, 0.f, 0.f, 0.f};
            acc1 = (v8f){0.f, 0.f, 0.f, 0.f, 0.f, 0.f, 0.f, 0.f};
        }
        acc0 = __builtin_amdgcn_wmma_f32_16x16x32_bf16(
            false, af[step & 1].v, false, bfrag[0][ec].v, (short)0, acc0, false, false);
        acc1 = __builtin_amdgcn_wmma_f32_16x16x32_bf16(
            false, af[step & 1].v, false, bfrag[1][ec].v, (short)0, acc1, false, false);
        if (ec == 3) {
            // C layout: VGPR r, lane: M = r + 8*hi, N = lo
            const float* ap = &amub[dc * 16 + hi * 8];
#pragma unroll
            for (int r = 0; r < 8; ++r) {
                const float t0 = acc0[r] - ap[r];
                const float t1 = acc1[r] - ap[r];
                sq0 = __builtin_fmaf(t0, t0, sq0);
                sq1 = __builtin_fmaf(t1, t1, sq1);
            }
        }
    }
    sq0o = sq0; sq1o = sq1;
}

// ---------------------------------------------------------------------------
// Main fused kernel: block = 256 n-columns x 2 consecutive c's.
//   sq[c*N + n] = sum_d ( (A_c X)[d,n] - Amu[d,c] )^2
// X tile + both A_c tiles delivered by TDM (async, LDS-double-buffered A).
// grid (N/256, K/2), block 256 (8 waves, wave -> 32 n-cols)
// ---------------------------------------------------------------------------
__global__ __launch_bounds__(256, 1)
void gemm_sq_kernel(const __bf16* __restrict__ Abf, const __bf16* __restrict__ Xt,
                    const float* __restrict__ Amu, float* __restrict__ sq) {
    __shared__ __bf16 lX[256 * PITCH];       // 256 n-rows of 128 e
    __shared__ __bf16 lA[2][DD * PITCH];     // double-buffered A_c
    __shared__ float  lAmu[2][DD];

    const int tid   = threadIdx.x;
    const int lane  = tid & 31;
    const int wave  = tid >> 5;
    const int lo    = lane & 15;
    const int hi    = lane >> 4;
    const int nBase = blockIdx.x * 256;
    const int c0    = blockIdx.y * 2;
    const bool tdmW = (wave == 0);

    // ---- kick off async TDM fills: X tile, then A(c0) ----
    if (tdmW) {
        tdm_load_tile(Xt + (size_t)nBase * DD,
                      (unsigned)(uintptr_t)&lX[0], 256);
        tdm_load_tile(Abf + (size_t)c0 * (DD * DD),
                      (unsigned)(uintptr_t)&lA[0][0], DD);
    }
    // Amu for both c's (tiny, vector path)
    lAmu[tid >> 7][tid & 127] = Amu[(size_t)(c0 + (tid >> 7)) * DD + (tid & 127)];

    if (tdmW) __builtin_amdgcn_s_wait_tensorcnt((short)1);   // X landed
    __syncthreads();

    // ---- preload B fragments: 2 strips x 4 K-chunks (register resident) ----
    Frag bfrag[2][4];
#pragma unroll
    for (int s = 0; s < 2; ++s)
#pragma unroll
        for (int ec = 0; ec < 4; ++ec)
            load_frag(bfrag[s][ec],
                      &lX[(wave * 32 + s * 16 + lo) * PITCH + ec * 32], hi);

    // issue A(c1) now; it streams in under the c0 compute
    if (tdmW) {
        tdm_load_tile(Abf + (size_t)(c0 + 1) * (DD * DD),
                      (unsigned)(uintptr_t)&lA[1][0], DD);
        __builtin_amdgcn_s_wait_tensorcnt((short)1);         // A(c0) landed
    }
    __syncthreads();

#pragma unroll
    for (int ci = 0; ci < 2; ++ci) {
        float sq0, sq1;
        compute_c(&lA[ci][0], &lAmu[ci][0], bfrag, lo, hi, sq0, sq1);
        // fold lane-halves (rows M and M+8 live in lane and lane^16)
        sq0 += __shfl_xor(sq0, 16, 32);
        sq1 += __shfl_xor(sq1, 16, 32);
        if (hi == 0) {
            const int n0 = nBase + wave * 32;
            sq[(size_t)(c0 + ci) * NN + n0 + lo]      = sq0;
            sq[(size_t)(c0 + ci) * NN + n0 + 16 + lo] = sq1;
        }
        if (ci == 0) {
            if (tdmW) __builtin_amdgcn_s_wait_tensorcnt((short)0);  // A(c1) landed
            __syncthreads();
        }
    }
}

// ---------------------------------------------------------------------------
// Softmax over c (K=256) per column n; sq is [c][n] so fully coalesced.
// ---------------------------------------------------------------------------
__global__ __launch_bounds__(256)
void softmax_kernel(const float* __restrict__ sq, const float* __restrict__ gammaPtr,
                    float* __restrict__ out) {
    const int n = blockIdx.x * blockDim.x + threadIdx.x;
    const float g = *gammaPtr;
    float m = -3.4e38f;
    for (int c = 0; c < KK; ++c) m = fmaxf(m, g * sq[(size_t)c * NN + n]);
    float s = 0.f;
    for (int c = 0; c < KK; ++c) s += __expf(g * sq[(size_t)c * NN + n] - m);
    const float inv = 1.f / s;
    for (int c = 0; c < KK; ++c)
        out[(size_t)(DD + c) * NN + n] = __expf(g * sq[(size_t)c * NN + n] - m) * inv;
}

// ---------------------------------------------------------------------------
// out rows [0,128) = X (fp32 passthrough), float4 copy
// ---------------------------------------------------------------------------
__global__ __launch_bounds__(256)
void copyx_kernel(const float4* __restrict__ in, float4* __restrict__ out) {
    const int i = blockIdx.x * blockDim.x + threadIdx.x;
    out[i] = in[i];
}

// ---------------------------------------------------------------------------
extern "C" void kernel_launch(void* const* d_in, const int* in_sizes, int n_in,
                              void* d_out, int out_size, void* d_ws, size_t ws_size,
                              hipStream_t stream) {
    const float* X     = (const float*)d_in[0];   // (128, 8192)
    const float* NA    = (const float*)d_in[1];   // (128, 128, 256)
    const float* Nmu   = (const float*)d_in[2];   // (128, 256)
    const float* gamma = (const float*)d_in[3];   // scalar
    float* out = (float*)d_out;                   // (384, 8192)

    char* ws = (char*)d_ws;
    __bf16* Abf = (__bf16*)(ws);                               //  8 MiB [c][d][e]
    __bf16* Xt  = (__bf16*)(ws + (size_t)(8u << 20));          //  2 MiB [n][e]
    float*  Amu = (float*) (ws + (size_t)(10u << 20));         // 128 KiB [c][d]
    float*  sqb = (float*) (ws + (size_t)(10u << 20) + (128u << 10)); // 8 MiB [c][n]

    // 1) A: (16384 x 256) fp32 -> (256 x 16384) bf16
    transpose_cvt_kernel<<<dim3(KK / 32, (DD * DD) / 32), dim3(32, 8), 0, stream>>>(
        NA, Abf, DD * DD, KK);
    // 2) X: (128 x 8192) fp32 -> (8192 x 128) bf16
    transpose_cvt_kernel<<<dim3(NN / 32, DD / 32), dim3(32, 8), 0, stream>>>(
        X, Xt, DD, NN);
    // 3) Amu
    amu_kernel<<<DD, KK, 0, stream>>>(NA, Nmu, Amu);
    // 4) batched WMMA GEMM + fused (.-Amu)^2 d-reduction (TDM-fed)
    gemm_sq_kernel<<<dim3(NN / 256, KK / 2), 256, 0, stream>>>(Abf, Xt, Amu, sqb);
    // 5) passthrough X rows
    copyx_kernel<<<(DD * NN / 4) / 256, 256, 0, stream>>>(
        (const float4*)X, (float4*)out);
    // 6) softmax over c -> Pi rows
    softmax_kernel<<<NN / 256, 256, 0, stream>>>(sqb, gamma, out);
}
